// GumbelVQTokenizer_80092550136306
// MI455X (gfx1250) — matrix operand order repaired and checked by reading, not compiled, not run on hardware
//
#include <hip/hip_runtime.h>
#include <cstddef>
#include <cstdint>

// ---------------------------------------------------------------------------
// Gumbel VQ tokenizer, MI455X (gfx1250, wave32, WMMA).
//
// Roofline: ~640 MB moved -> ~27.5 us floor @ 23.3 TB/s HBM; 34.4 GFLOP of
// GEMM is negligible in bf16 WMMA (16x16x32, f32 accum) but would not be in
// f32 WMMA (16x16x4). => memory-bound kernel built around bf16 WMMA.
//
// d_ws layout (bf16): [0,65536)  codebook^T in GEMM1 B-fragment layout
//                     [65536,131072) codebook in GEMM2 B-fragment layout
// B-frag layout used here (contiguous K, same order as the A loads): lane L holds N = L&15, elements i=0..15 hold
// K = (L<16 ? i : 16+i) + 32*kstep, stored contiguously (32B per lane).
// ---------------------------------------------------------------------------

typedef __attribute__((ext_vector_type(16))) __bf16 v16bf;
typedef __attribute__((ext_vector_type(8)))  __bf16 v8bf;
typedef __attribute__((ext_vector_type(4)))  __bf16 v4bf;
typedef __attribute__((ext_vector_type(8)))  float  v8f;
typedef __attribute__((ext_vector_type(4)))  float  v4f;   // plain vector, OK for NT builtins

#define DIM   128
#define CODES 512
#define CB2_OFF 65536   // element offset of GEMM2 codebook frags in ws

union FragBF { v16bf v; v8bf h[2]; };

// ---------------------------------------------------------------------------
// Prep: swizzle f32 codebook into the two bf16 WMMA B-fragment layouts.
// ---------------------------------------------------------------------------
__global__ void __launch_bounds__(256)
vq_prep(const float* __restrict__ cb, __bf16* __restrict__ ws)
{
    const int stride = gridDim.x * blockDim.x;
    for (int e = blockIdx.x * blockDim.x + threadIdx.x; e < 2 * 65536; e += stride) {
        int val_n, val_k;
        int idx = e;
        if (e < 65536) {
            // GEMM1: B = codebook^T, frag (ks in 0..3, t in 0..31)
            const int i    = e & 15;
            const int lane = (e >> 4) & 31;
            const int f    = e >> 9;
            const int t    = f & 31;
            const int ks   = f >> 5;
            val_n = t * 16 + (lane & 15);                      // code row of cb
            val_k = ks * 32 + ((lane < 16) ? i : 16 + i);      // feature
            ws[idx] = (__bf16)cb[val_n * DIM + val_k];
        } else {
            // GEMM2: B = codebook, frag (ks in 0..15, t in 0..7)
            const int e2   = e - 65536;
            const int i    = e2 & 15;
            const int lane = (e2 >> 4) & 31;
            const int f    = e2 >> 9;
            const int t    = f & 7;
            const int ks   = f >> 3;
            val_k = ks * 32 + ((lane < 16) ? i : 16 + i);      // code row of cb
            val_n = t * 16 + (lane & 15);                      // feature
            ws[idx] = (__bf16)cb[val_k * DIM + val_n];
        }
    }
}

// ---------------------------------------------------------------------------
// Main: 8 waves/block, each wave owns 16 rows end-to-end.
// Dynamic LDS: 8 waves * 16KB (xstage[16][128]bf16 then probs[16][512]bf16,
// non-overlapping lifetimes in the same slab).
// ---------------------------------------------------------------------------
__global__ void __launch_bounds__(256)
vq_main(const float* __restrict__ x,
        const float* __restrict__ mask,
        const float* __restrict__ noise,
        const __bf16* __restrict__ cbfrag,
        float* __restrict__ outq,
        float* __restrict__ oute,
        float* __restrict__ outi)
{
    extern __shared__ __bf16 smem[];
    const int tid      = threadIdx.x;
    const int wave     = tid >> 5;
    const int lane     = tid & 31;
    const int lane15   = lane & 15;
    const int laneHalf = lane >> 4;          // 0: lanes 0-15, 1: lanes 16-31
    const int loOff    = laneHalf * 16;      // A-frag K sub-offset per half
    __bf16* wbuf = smem + wave * 8192;       // 16KB per-wave slab
    const int rowBase = blockIdx.x * 128 + wave * 16;

    // ---- Phase 1: load x, mask-nudge, L2-normalize, stage bf16 to LDS ----
    #pragma unroll
    for (int r = 0; r < 16; ++r) {
        const int row = rowBase + r;
        const v4f f =
            __builtin_nontemporal_load((const v4f*)(x + (size_t)row * DIM + lane * 4));
        const float nm = (1.0f - mask[row]) * 1e-6f;
        const float a0 = f.x + nm, a1 = f.y + nm, a2 = f.z + nm, a3 = f.w + nm;
        float ss = a0 * a0 + a1 * a1 + a2 * a2 + a3 * a3;
        #pragma unroll
        for (int d = 1; d < 32; d <<= 1) ss += __shfl_xor(ss, d, 32);
        const float scale = 1.0f / fmaxf(sqrtf(ss), 1e-6f);
        v4bf o;
        o[0] = (__bf16)(a0 * scale);
        o[1] = (__bf16)(a1 * scale);
        o[2] = (__bf16)(a2 * scale);
        o[3] = (__bf16)(a3 * scale);
        *(v4bf*)(wbuf + r * DIM + lane * 4) = o;
        // Warm the noise row (2KB = 32 lanes * 64B) -> global_prefetch_b8
        __builtin_prefetch(noise + (size_t)row * CODES + lane * 16, 0, 1);
    }
    // Same-wave LDS RAW: compiler inserts s_wait_dscnt; no barrier needed.

    // ---- Phase 2: GEMM1  ab(16x512) = xn(16x128) @ cb^T(128x512) ----
    v8f acc[32];
    #pragma unroll
    for (int t = 0; t < 32; ++t)
        #pragma unroll
        for (int e = 0; e < 8; ++e) acc[t][e] = 0.0f;

    #pragma unroll
    for (int ks = 0; ks < 4; ++ks) {
        FragBF A;
        const __bf16* ap = wbuf + lane15 * DIM + ks * 32 + loOff;
        A.h[0] = *(const v8bf*)ap;            // K = 32ks + loOff + {0..7}
        A.h[1] = *(const v8bf*)(ap + 8);      // K = 32ks + loOff + {8..15}
        #pragma unroll
        for (int t = 0; t < 32; ++t) {
            FragBF B;
            const __bf16* bp = cbfrag + ((size_t)(ks * 32 + t) * 512 + lane * 16);
            B.h[0] = *(const v8bf*)bp;
            B.h[1] = *(const v8bf*)(bp + 8);
            acc[t] = __builtin_amdgcn_wmma_f32_16x16x32_bf16(
                false, A.v, false, B.v, (short)0, acc[t], false, false);
        }
    }

    // ---- Phase 3: fused argmax + gumbel-softmax on the WMMA C layout ----
    // acc[t][r] = ab[row = r + 8*laneHalf][col = 16t + lane15]
    #pragma unroll
    for (int r = 0; r < 8; ++r) {
        const int rowLoc = laneHalf * 8 + r;
        const int row    = rowBase + rowLoc;
        float lg[32];
        float m = -3.0e38f, bestAb = -3.0e38f;
        int bestIdx = 0;
        #pragma unroll
        for (int t = 0; t < 32; ++t) {
            const float ab = acc[t][r];
            const float ns = __builtin_nontemporal_load(
                noise + (size_t)row * CODES + 16 * t + lane15);
            const float l = 2.0f * ab + ns;   // == -dist + noise + const(row)
            lg[t] = l;
            m = fmaxf(m, l);
            if (ab > bestAb) { bestAb = ab; bestIdx = 16 * t + lane15; }
        }
        // reductions over the 16 lanes holding this row (xor<16 stays in half)
        #pragma unroll
        for (int d = 1; d < 16; d <<= 1) m = fmaxf(m, __shfl_xor(m, d, 32));
        #pragma unroll
        for (int d = 1; d < 16; d <<= 1) {
            const float oa = __shfl_xor(bestAb, d, 32);
            const int   oi = __shfl_xor(bestIdx, d, 32);
            if (oa > bestAb || (oa == bestAb && oi < bestIdx)) { bestAb = oa; bestIdx = oi; }
        }
        float s = 0.0f;
        #pragma unroll
        for (int t = 0; t < 32; ++t) { const float e = __expf(lg[t] - m); lg[t] = e; s += e; }
        #pragma unroll
        for (int d = 1; d < 16; d <<= 1) s += __shfl_xor(s, d, 32);
        const float rs = 1.0f / s;
        #pragma unroll
        for (int t = 0; t < 32; ++t) {
            const float p = lg[t] * rs;
            __builtin_nontemporal_store(p, oute + (size_t)row * CODES + 16 * t + lane15);
            wbuf[rowLoc * CODES + 16 * t + lane15] = (__bf16)p;  // A for GEMM2
        }
        if (lane15 == 0) outi[row] = (float)bestIdx;   // argmin(dist) = argmax(ab)
    }

    // ---- Phase 4: GEMM2  q(16x128) = P(16x512) @ cb(512x128) ----
    v8f acc2[8];
    #pragma unroll
    for (int t = 0; t < 8; ++t)
        #pragma unroll
        for (int e = 0; e < 8; ++e) acc2[t][e] = 0.0f;

    #pragma unroll
    for (int ks = 0; ks < 16; ++ks) {
        FragBF A;
        const __bf16* ap = wbuf + lane15 * CODES + ks * 32 + loOff;
        A.h[0] = *(const v8bf*)ap;
        A.h[1] = *(const v8bf*)(ap + 8);
        #pragma unroll
        for (int t = 0; t < 8; ++t) {
            FragBF B;
            const __bf16* bp = cbfrag + CB2_OFF + ((size_t)(ks * 8 + t) * 512 + lane * 16);
            B.h[0] = *(const v8bf*)bp;
            B.h[1] = *(const v8bf*)(bp + 8);
            acc2[t] = __builtin_amdgcn_wmma_f32_16x16x32_bf16(
                false, A.v, false, B.v, (short)0, acc2[t], false, false);
        }
    }
    #pragma unroll
    for (int t = 0; t < 8; ++t)
        #pragma unroll
        for (int r = 0; r < 8; ++r)
            __builtin_nontemporal_store(
                acc2[t][r],
                outq + (size_t)(rowBase + laneHalf * 8 + r) * DIM + 16 * t + lane15);
}

// ---------------------------------------------------------------------------
extern "C" void kernel_launch(void* const* d_in, const int* in_sizes, int n_in,
                              void* d_out, int out_size, void* d_ws, size_t ws_size,
                              hipStream_t stream)
{
    (void)n_in; (void)out_size; (void)ws_size;  // ws needs 256KB (bf16 frags)
    const float* x     = (const float*)d_in[0];
    const float* mask  = (const float*)d_in[1];
    const float* cb    = (const float*)d_in[2];
    const float* noise = (const float*)d_in[3];

    const int n = in_sizes[0] / DIM;            // 131072 rows
    float* out  = (float*)d_out;
    float* outq = out;                          // (n,128)
    float* oute = out + (size_t)n * DIM;        // (n,512)
    float* outi = out + (size_t)n * (DIM + CODES); // (n,) indices as float

    __bf16* ws = (__bf16*)d_ws;

    // 1) swizzle codebook into WMMA B-fragment layouts (both GEMMs)
    vq_prep<<<128, 256, 0, stream>>>(cb, ws);

    // 2) fused normalize -> GEMM1 -> softmax/argmax -> GEMM2
    const int blocks = n / 128;                 // 128 rows per 8-wave block
    vq_main<<<blocks, 256, 8 * 8192 * (int)sizeof(__bf16), stream>>>(
        x, mask, noise, ws, outq, oute, outi);
}
